// AnomalyDAEBase_59253368816204
// MI455X (gfx1250) — compile-verified, matrix operand order checked by reading, not run on hardware
//
#include <hip/hip_runtime.h>

typedef _Float16 half_t;
typedef __attribute__((ext_vector_type(16))) _Float16 v16h;
typedef __attribute__((ext_vector_type(8)))  _Float16 v8h;
typedef __attribute__((ext_vector_type(8)))  float    v8f;

#define NN   10000
#define EE   320000
#define IND  512
#define EMBD 256
#define HIDD 128
#define KPAD 10016      // K=10000 padded to multiple of 32
#define NEG_SLOPE 0.2f

// ---------- helpers: monotone uint encoding of float for atomicMax ----------
__device__ __forceinline__ unsigned f2ord(float f) {
  unsigned u = __float_as_uint(f);
  return (u & 0x80000000u) ? ~u : (u | 0x80000000u);
}
__device__ __forceinline__ float ord2f(unsigned u) {
  return (u & 0x80000000u) ? __uint_as_float(u & 0x7fffffffu)
                           : __uint_as_float(~u);
}

// ---------- elementwise f32 -> f16 ----------
__global__ void k_f32_to_f16(half_t* __restrict__ dst, const float* __restrict__ src, int n) {
  int i = blockIdx.x * blockDim.x + threadIdx.x;
  if (i < n) dst[i] = (half_t)src[i];
}

// ---------- transpose + convert: src[R,C] f32 -> dst[C,Ld] f16, zero-pad r >= R ----------
__global__ void k_transpose_f16(half_t* __restrict__ dst, const float* __restrict__ src,
                                int R, int C, int Ld) {
  long i = (long)blockIdx.x * blockDim.x + threadIdx.x;
  long total = (long)C * Ld;
  if (i >= total) return;
  int c = (int)(i / Ld);
  int r = (int)(i - (long)c * Ld);
  dst[i] = (r < R) ? (half_t)src[(size_t)r * C + c] : (half_t)0.0f;
}

// ---------- WMMA GEMM: C[M,N] = act(A[M,K] * B[N,K]^T + bias) ----------
// A,B f16 row-major; one wave computes a 16 x (16*NB) tile. Per K-step the A
// fragment and ALL NB B fragments are loaded first (single load clause, partial
// s_wait_loadcnt overlap), then the NB WMMAs issue back-to-back reusing A.
// K multiple of 32; N/16 multiple of NB.  MODE: 0 = identity, 1 = relu, 2 = sigmoid
template<int MODE, int NB>
__global__ void __launch_bounds__(32)
k_wmma_gemm_nt(const half_t* __restrict__ A, const half_t* __restrict__ B,
               int K, int lda, int ldb, int ldc,
               const float* __restrict__ bias,
               float* __restrict__ Cf, half_t* __restrict__ Ch)
{
  const int lane = threadIdx.x;
  const int r    = lane & 15;   // M-row for A frag / N-col for B frags
  const int hi   = lane >> 4;   // half-wave selects K sub-block
  const int m0   = blockIdx.y << 4;
  const int n0   = blockIdx.x * (16 * NB);

  const half_t* Arow = A + (size_t)(m0 + r) * lda;
  const half_t* Brow[NB];
#pragma unroll
  for (int j = 0; j < NB; ++j)
    Brow[j] = B + (size_t)(n0 + j * 16 + r) * ldb;

  v8f acc[NB];
#pragma unroll
  for (int j = 0; j < NB; ++j) acc[j] = (v8f){};

  for (int k0 = 0; k0 < K; k0 += 32) {
    const int kb = k0 + hi * 8;                 // 16B-aligned K offset per half-wave
    union { v16h v; v8h h[2]; } a, b[NB];
    // --- issue all loads first so they form one clause and overlap the WMMAs
    a.h[0] = *(const v8h*)(Arow + kb);          // K = kb .. kb+7
    a.h[1] = *(const v8h*)(Arow + kb + 16);     // K = kb+16 .. kb+23
#pragma unroll
    for (int j = 0; j < NB; ++j) {
      b[j].h[0] = *(const v8h*)(Brow[j] + kb);
      b[j].h[1] = *(const v8h*)(Brow[j] + kb + 16);
    }
    // --- then the math, reusing the A fragment NB times
#pragma unroll
    for (int j = 0; j < NB; ++j)
      acc[j] = __builtin_amdgcn_wmma_f32_16x16x32_f16(false, a.v, false, b[j].v,
                                                      (short)0, acc[j], false, false);
  }

#pragma unroll
  for (int j = 0; j < NB; ++j) {
    const int col = n0 + j * 16 + r;
    const float bv = bias ? bias[col] : 0.0f;
#pragma unroll
    for (int i = 0; i < 8; ++i) {
      const int row = m0 + hi * 8 + i;          // C/D layout: VGPR i -> M = i + 8*(lane/16)
      float v = acc[j][i] + bv;
      if (MODE == 1) v = (v > 0.0f) ? v : 0.0f;
      if (MODE == 2) v = 1.0f / (1.0f + __expf(-v));
      if (Cf) Cf[(size_t)row * ldc + col] = v;
      if (Ch) Ch[(size_t)row * ldc + col] = (half_t)v;
    }
  }
}

// ---------- GAT scalar kernels ----------
__global__ void k_node_dots(const float* __restrict__ Wh, const float* __restrict__ ws,
                            const float* __restrict__ wd, float* __restrict__ a_s,
                            float* __restrict__ a_d, int n) {
  int i = blockIdx.x * blockDim.x + threadIdx.x;
  if (i >= n) return;
  const float* row = Wh + (size_t)i * HIDD;
  float s = 0.f, d = 0.f;
  for (int k = 0; k < HIDD; ++k) { float v = row[k]; s += v * ws[k]; d += v * wd[k]; }
  a_s[i] = s; a_d[i] = d;
}

__global__ void k_gat_init(unsigned* __restrict__ emax, float* __restrict__ denom,
                           float* __restrict__ h2acc, int n, int nh) {
  int i = blockIdx.x * blockDim.x + threadIdx.x;
  if (i < n) { emax[i] = f2ord(-__builtin_inff()); denom[i] = 0.f; }
  if (i < nh) h2acc[i] = 0.f;
}

__global__ void k_edge_logits(const int* __restrict__ ei, const float* __restrict__ a_s,
                              const float* __restrict__ a_d, float* __restrict__ el,
                              unsigned* __restrict__ emax, int E, int tot) {
  int e = blockIdx.x * blockDim.x + threadIdx.x;
  if (e >= tot) return;
  int s = (e < E) ? ei[e]     : (e - E);   // self-loops appended
  int d = (e < E) ? ei[E + e] : (e - E);
  float v = a_s[s] + a_d[d];
  v = (v > 0.f) ? v : NEG_SLOPE * v;       // leaky_relu
  el[e] = v;
  atomicMax(&emax[d], f2ord(v));
}

__global__ void k_edge_exp(const int* __restrict__ ei, float* __restrict__ el,
                           const unsigned* __restrict__ emax, float* __restrict__ denom,
                           int E, int tot) {
  int e = blockIdx.x * blockDim.x + threadIdx.x;
  if (e >= tot) return;
  int d = (e < E) ? ei[E + e] : (e - E);
  float w = __expf(el[e] - ord2f(emax[d]));
  el[e] = w;
  atomicAdd(&denom[d], w);
}

__global__ void k_edge_scatter(const int* __restrict__ ei, const float* __restrict__ el,
                               const float* __restrict__ denom, const float* __restrict__ Wh,
                               float* __restrict__ h2acc, int E, int tot) {
  long i = (long)blockIdx.x * blockDim.x + threadIdx.x;
  long total = (long)tot * HIDD;
  if (i >= total) return;
  int e = (int)(i >> 7);          // HIDD == 128
  int k = (int)(i & (HIDD - 1));
  int s = (e < E) ? ei[e]     : (e - E);
  int d = (e < E) ? ei[E + e] : (e - E);
  float alpha = el[e] / denom[d];
  atomicAdd(&h2acc[(size_t)d * HIDD + k], alpha * Wh[(size_t)s * HIDD + k]);
}

__global__ void k_h2_final(const float* __restrict__ h2acc, const float* __restrict__ b,
                           half_t* __restrict__ h2_16, int nh) {
  int i = blockIdx.x * blockDim.x + threadIdx.x;
  if (i >= nh) return;
  h2_16[i] = (half_t)(h2acc[i] + b[i & (HIDD - 1)]);
}

// =====================================================================
extern "C" void kernel_launch(void* const* d_in, const int* in_sizes, int n_in,
                              void* d_out, int out_size, void* d_ws, size_t ws_size,
                              hipStream_t stream) {
  const float* x      = (const float*)d_in[0];
  const int*   ei     = (const int*)  d_in[1];
  // d_in[2] = batch_size (== NN, unused)
  const float* W_stru = (const float*)d_in[3];
  const float* b_stru = (const float*)d_in[4];
  const float* W_gat  = (const float*)d_in[5];
  const float* att_s  = (const float*)d_in[6];
  const float* att_d  = (const float*)d_in[7];
  const float* b_gat  = (const float*)d_in[8];
  const float* W_a1   = (const float*)d_in[9];
  const float* b_a1   = (const float*)d_in[10];
  const float* W_a2   = (const float*)d_in[11];
  const float* b_a2   = (const float*)d_in[12];

  float* out_x = (float*)d_out;                       // [NN, IND]
  float* out_s = (float*)d_out + (size_t)NN * IND;    // [NN, NN]

  // ---- workspace carve (256B aligned) ----
  char* p = (char*)d_ws;
  auto carve = [&](size_t bytes) -> void* {
    void* r = (void*)p;
    p += (bytes + 255) & ~(size_t)255;
    return r;
  };
  half_t*   x16    = (half_t*)  carve((size_t)NN   * IND  * 2);
  half_t*   WstruT = (half_t*)  carve((size_t)EMBD * IND  * 2);
  half_t*   h16    = (half_t*)  carve((size_t)NN   * EMBD * 2);
  half_t*   WgatT  = (half_t*)  carve((size_t)HIDD * EMBD * 2);
  float*    Wh     = (float*)   carve((size_t)NN   * HIDD * 4);
  float*    a_src  = (float*)   carve((size_t)NN * 4);
  float*    a_dst  = (float*)   carve((size_t)NN * 4);
  unsigned* emax   = (unsigned*)carve((size_t)NN * 4);
  float*    denom  = (float*)   carve((size_t)NN * 4);
  float*    el     = (float*)   carve((size_t)(EE + NN) * 4);
  float*    h2acc  = (float*)   carve((size_t)NN   * HIDD * 4);
  half_t*   h2_16  = (half_t*)  carve((size_t)NN   * HIDD * 2);
  half_t*   xT16   = (half_t*)  carve((size_t)IND  * KPAD * 2);
  half_t*   Wa1T   = (half_t*)  carve((size_t)EMBD * KPAD * 2);
  half_t*   xa1_16 = (half_t*)  carve((size_t)IND  * EMBD * 2);
  half_t*   Wa2T   = (half_t*)  carve((size_t)HIDD * EMBD * 2);
  half_t*   xa16   = (half_t*)  carve((size_t)IND  * HIDD * 2);

  // ---- precision conversions / transposes ----
  { int n = NN * IND;
    k_f32_to_f16<<<(n + 255) / 256, 256, 0, stream>>>(x16, x, n); }
  { long t = (long)EMBD * IND;
    k_transpose_f16<<<(unsigned)((t + 255) / 256), 256, 0, stream>>>(WstruT, W_stru, IND, EMBD, IND); }
  { long t = (long)HIDD * EMBD;
    k_transpose_f16<<<(unsigned)((t + 255) / 256), 256, 0, stream>>>(WgatT, W_gat, EMBD, HIDD, EMBD); }
  { long t = (long)IND * KPAD;
    k_transpose_f16<<<(unsigned)((t + 255) / 256), 256, 0, stream>>>(xT16, x, NN, IND, KPAD); }
  { long t = (long)EMBD * KPAD;
    k_transpose_f16<<<(unsigned)((t + 255) / 256), 256, 0, stream>>>(Wa1T, W_a1, NN, EMBD, KPAD); }
  { long t = (long)HIDD * EMBD;
    k_transpose_f16<<<(unsigned)((t + 255) / 256), 256, 0, stream>>>(Wa2T, W_a2, EMBD, HIDD, EMBD); }

  const dim3 blk(32);

  // ---- structure encoder: h = relu(x @ W_stru + b_stru) -> h16 [NN,EMBD]
  k_wmma_gemm_nt<1, 4><<<dim3(EMBD / 64, NN / 16), blk, 0, stream>>>(
      x16, WstruT, IND, IND, IND, EMBD, b_stru, nullptr, h16);

  // ---- GAT: Wh = h @ W_gat -> f32 [NN,HIDD]
  k_wmma_gemm_nt<0, 4><<<dim3(HIDD / 64, NN / 16), blk, 0, stream>>>(
      h16, WgatT, EMBD, EMBD, EMBD, HIDD, nullptr, Wh, nullptr);

  k_node_dots<<<(NN + 255) / 256, 256, 0, stream>>>(Wh, att_s, att_d, a_src, a_dst, NN);
  k_gat_init<<<(NN * HIDD + 255) / 256, 256, 0, stream>>>(emax, denom, h2acc, NN, NN * HIDD);

  const int tot = EE + NN;
  k_edge_logits<<<(tot + 255) / 256, 256, 0, stream>>>(ei, a_src, a_dst, el, emax, EE, tot);
  k_edge_exp   <<<(tot + 255) / 256, 256, 0, stream>>>(ei, el, emax, denom, EE, tot);
  { long st = (long)tot * HIDD;
    k_edge_scatter<<<(unsigned)((st + 255) / 256), 256, 0, stream>>>(ei, el, denom, Wh, h2acc, EE, tot); }
  k_h2_final<<<(NN * HIDD + 255) / 256, 256, 0, stream>>>(h2acc, b_gat, h2_16, NN * HIDD);

  // ---- structure recon: s_ = sigmoid(h2 @ h2^T) -> out_s [NN,NN]
  // dominant stage: 400MB stores; NB=5 (625 = 5*125) reuses each A fragment 5x
  k_wmma_gemm_nt<2, 5><<<dim3(NN / 80, NN / 16), blk, 0, stream>>>(
      h2_16, h2_16, HIDD, HIDD, HIDD, NN, nullptr, out_s, nullptr);

  // ---- attribute encoder: xa1 = relu(x^T @ W_a1 + b_a1) -> [IND,EMBD]
  k_wmma_gemm_nt<1, 4><<<dim3(EMBD / 64, IND / 16), blk, 0, stream>>>(
      xT16, Wa1T, KPAD, KPAD, KPAD, EMBD, b_a1, nullptr, xa1_16);

  // ---- xa = xa1 @ W_a2 + b_a2 -> [IND,HIDD]
  k_wmma_gemm_nt<0, 4><<<dim3(HIDD / 64, IND / 16), blk, 0, stream>>>(
      xa1_16, Wa2T, EMBD, EMBD, EMBD, HIDD, b_a2, nullptr, xa16);

  // ---- attribute recon: x_ = h2 @ xa^T -> out_x [NN,IND]
  k_wmma_gemm_nt<0, 4><<<dim3(IND / 64, NN / 16), blk, 0, stream>>>(
      h2_16, xa16, HIDD, HIDD, HIDD, IND, nullptr, out_x, nullptr);
}
